// EqPropRobot_88175678587612
// MI455X (gfx1250) — matrix operand order repaired and checked by reading, not compiled, not run on hardware
//
#include <hip/hip_runtime.h>
#include <hip/hip_bf16.h>

// Problem dims (fixed by the reference)
#define BB      4096
#define IN_DIM  1024
#define HH      4096
#define OUT_DIM 1024
#define T_RELAX 30

typedef __attribute__((ext_vector_type(16))) __bf16 v16bf;
typedef __attribute__((ext_vector_type(8)))  float  v8f;

// GEMM tiling: 256x128 block tile, 32-K step, 8 waves (wave32) per block,
// each wave = 64x64 sub-tile = 4x4 WMMA 16x16 tiles (16 WMMA / K-step,
// 1.0 ds_load_b128 per WMMA).
constexpr int BM = 256, BN = 128, BK = 32;
constexpr int KPAD = 40; // LDS K-stride in bf16 elems (80B rows -> bank spread, 16B aligned)

constexpr float EPS_  = 0.5f;
constexpr float BETA_ = 0.5f;
constexpr float CH_ = 1.0f - EPS_;          // h_old coefficient = 0.5
constexpr float CT_ = 2.0f * BETA_ * EPS_;  // target coefficient = 0.5
// y_old coefficient = 1 - EPS - 2*BETA*EPS == 0.0 exactly -> y_old never read.

union Frag16 {
  v16bf v;
  unsigned int u[8];
  uint4 q[2];
};

// C = A[MxK] * Bn[NxK]^T  with fused epilogue.
// MODE 0: out_f32 = C + bias                              (xW1b precompute)
// MODE 1: h' = clip(CH*h_old + EPS*(xW1b + C)); f32+bf16  (h update)
// MODE 2: y' = clip(EPS*(C + bias) + CT*target); f32+bf16 (y update)
template<int MODE>
__global__ __launch_bounds__(256)
void gemm_bf16_wmma(const __bf16* __restrict__ A,   // [M][K] row-major bf16
                    const __bf16* __restrict__ Bn,  // [N][K] row-major bf16 (i.e. B^T)
                    int M, int N, int K,
                    const float* __restrict__ bias,
                    const float* __restrict__ add_f32,
                    const float* __restrict__ old_f32,
                    const float* __restrict__ target,
                    float* __restrict__ out_f32,
                    __bf16* __restrict__ out_bf)
{
  __shared__ __bf16 As[BM * KPAD];   // 256 x 32 (padded to 40)
  __shared__ __bf16 Bs[BN * KPAD];   // 128 x 32 (padded to 40)

  const int m0   = blockIdx.y * BM;
  const int n0   = blockIdx.x * BN;
  const int tid  = threadIdx.x;
  const int lane = tid & 31;
  const int wave = tid >> 5;
  const int l16  = lane & 15;
  const int lhi  = lane >> 4;        // half-wave select
  const int wm   = (wave & 3) * 64;  // 4 waves along M
  const int wn   = (wave >> 2) * 64; // 2 waves along N

  v8f acc[4][4] = {};

  for (int k0 = 0; k0 < K; k0 += BK) {
    // Global -> LDS, fully coalesced b128s.
    // A tile: 256x32 bf16 = 1024 uint4 chunks (4/thread);
    // B tile: 128x32 bf16 =  512 uint4 chunks (2/thread).
    #pragma unroll
    for (int it = 0; it < 4; ++it) {
      int cc  = tid + it * 256;
      int row = cc >> 2;
      int qc  = (cc & 3) * 8;
      *(uint4*)(&As[row * KPAD + qc]) =
          *(const uint4*)(A + (long long)(m0 + row) * K + k0 + qc);
    }
    #pragma unroll
    for (int it = 0; it < 2; ++it) {
      int cc  = tid + it * 256;
      int row = cc >> 2;
      int qc  = (cc & 3) * 8;
      *(uint4*)(&Bs[row * KPAD + qc]) =
          *(const uint4*)(Bn + (long long)(n0 + row) * K + k0 + qc);
    }
    __syncthreads();

    // A fragments (16x32 bf16, ISA layout): lane holds row M=lane%16; VGPR j
    // holds K pair {0,2,4,6,16,18,20,22}[j] (+8 for lanes 16..31).
    // Elements 0..7 and 8..15 are each K-contiguous -> two b128 LDS loads.
    Frag16 af[4];
    #pragma unroll
    for (int mi = 0; mi < 4; ++mi) {
      const __bf16* ap = &As[(wm + mi * 16 + l16) * KPAD + lhi * 8];
      #pragma unroll
      for (int j = 0; j < 8; ++j) {
        int kk = (j < 4) ? (2 * j) : (16 + 2 * (j - 4));
        af[mi].u[j] = *(const unsigned int*)(ap + kk);
      }
    }
    // B fragments (32x16 bf16): lane holds column N=lane%16, K 0..7,16..23
    // (+8 for lanes 16..31), same order as A -> two b128 LDS loads.
    Frag16 bfr[4];
    #pragma unroll
    for (int ni = 0; ni < 4; ++ni) {
      const __bf16* bp = &Bs[(wn + ni * 16 + l16) * KPAD + lhi * 8];
      bfr[ni].q[0] = *(const uint4*)(bp);
      bfr[ni].q[1] = *(const uint4*)(bp + 16);
    }

    #pragma unroll
    for (int mi = 0; mi < 4; ++mi)
      #pragma unroll
      for (int ni = 0; ni < 4; ++ni)
        acc[mi][ni] = __builtin_amdgcn_wmma_f32_16x16x32_bf16(
            false, af[mi].v, false, bfr[ni].v, (short)0, acc[mi][ni],
            false, false);
    __syncthreads();
  }

  // Epilogue. C/D layout: VGPR r -> row r (+8 for lanes 16..31), col = lane%16.
  #pragma unroll
  for (int mi = 0; mi < 4; ++mi) {
    #pragma unroll
    for (int ni = 0; ni < 4; ++ni) {
      const int rbase = m0 + wm + mi * 16 + lhi * 8;
      const int col   = n0 + wn + ni * 16 + l16;
      #pragma unroll
      for (int r = 0; r < 8; ++r) {
        const long long idx = (long long)(rbase + r) * N + col;
        const float s = acc[mi][ni][r];
        if (MODE == 0) {
          out_f32[idx] = s + bias[col];
        } else if (MODE == 1) {
          float v = CH_ * old_f32[idx] + EPS_ * (add_f32[idx] + s);
          v = fminf(fmaxf(v, 0.0f), 1.0f);
          out_f32[idx] = v;
          out_bf[idx]  = (__bf16)v;
        } else {
          float v = EPS_ * (s + bias[col]) + CT_ * target[idx];
          v = fminf(fmaxf(v, 0.0f), 1.0f);
          out_f32[idx] = v;
          out_bf[idx]  = (__bf16)v;
        }
      }
    }
  }
}

__global__ __launch_bounds__(256)
void cvt_to_bf16(const float* __restrict__ in, __bf16* __restrict__ out, long long n) {
  long long i = (long long)blockIdx.x * blockDim.x + threadIdx.x;
  const long long stride = (long long)gridDim.x * blockDim.x;
  for (; i < n; i += stride) out[i] = (__bf16)in[i];
}

// out[c][r] = (bf16) in[r][c];  R, C multiples of 32.
__global__ __launch_bounds__(256)
void transpose_to_bf16(const float* __restrict__ in, __bf16* __restrict__ out,
                       int R, int C) {
  __shared__ float tile[32][33];
  const int bx = blockIdx.x * 32;  // over C
  const int by = blockIdx.y * 32;  // over R
  const int tx = threadIdx.x & 31;
  const int ty = threadIdx.x >> 5;
  #pragma unroll
  for (int i = 0; i < 32; i += 8)
    tile[ty + i][tx] = in[(long long)(by + ty + i) * C + (bx + tx)];
  __syncthreads();
  #pragma unroll
  for (int i = 0; i < 32; i += 8)
    out[(long long)(bx + ty + i) * R + (by + tx)] = (__bf16)tile[tx][ty + i];
}

__global__ __launch_bounds__(256)
void zero_u32(unsigned int* __restrict__ p, long long n) {
  long long i = (long long)blockIdx.x * blockDim.x + threadIdx.x;
  const long long stride = (long long)gridDim.x * blockDim.x;
  for (; i < n; i += stride) p[i] = 0u;
}

extern "C" void kernel_launch(void* const* d_in, const int* in_sizes, int n_in,
                              void* d_out, int out_size, void* d_ws, size_t ws_size,
                              hipStream_t stream) {
  const float* x   = (const float*)d_in[0];
  const float* W1  = (const float*)d_in[1];
  const float* b1  = (const float*)d_in[2];
  const float* W2  = (const float*)d_in[3];
  const float* b2  = (const float*)d_in[4];
  const float* tgt = (const float*)d_in[5];

  char* w = (char*)d_ws;
  auto carve = [&](size_t bytes) -> char* {
    char* p = w;
    w += (bytes + 255) & ~(size_t)255;
    return p;
  };
  __bf16* x_bf  = (__bf16*)carve((size_t)BB * IN_DIM * 2);       // x bf16 [B][IN]
  __bf16* W1t   = (__bf16*)carve((size_t)HH * IN_DIM * 2);       // W1^T bf16 [H][IN]
  __bf16* W2bf  = (__bf16*)carve((size_t)HH * OUT_DIM * 2);      // W2 bf16 [H][OUT] (N-major for y@W2^T)
  __bf16* W2t   = (__bf16*)carve((size_t)OUT_DIM * HH * 2);      // W2^T bf16 [OUT][H] (N-major for h@W2)
  float*  xW1b  = (float*) carve((size_t)BB * HH * 4);           // x@W1+b1
  float*  h_f32 = (float*) carve((size_t)BB * HH * 4);           // h state (updated in place)
  __bf16* h_bf  = (__bf16*)carve((size_t)BB * HH * 2);           // h bf16
  __bf16* y_bf0 = (__bf16*)carve((size_t)BB * OUT_DIM * 2);      // y bf16 ping
  __bf16* y_bf1 = (__bf16*)carve((size_t)BB * OUT_DIM * 2);      // y bf16 pong

  float* h_out = (float*)d_out;
  float* y_out = (float*)d_out + (size_t)BB * HH;

  // One-time prep: bf16 conversions / transposes of loop-invariant operands.
  cvt_to_bf16<<<2048, 256, 0, stream>>>(x,  x_bf,  (long long)BB * IN_DIM);
  cvt_to_bf16<<<2048, 256, 0, stream>>>(W2, W2bf, (long long)HH * OUT_DIM);
  transpose_to_bf16<<<dim3(HH / 32, IN_DIM / 32), 256, 0, stream>>>(W1, W1t, IN_DIM, HH);
  transpose_to_bf16<<<dim3(OUT_DIM / 32, HH / 32), 256, 0, stream>>>(W2, W2t, HH, OUT_DIM);
  zero_u32<<<2048, 256, 0, stream>>>((unsigned int*)h_f32, (long long)BB * HH);
  zero_u32<<<2048, 256, 0, stream>>>((unsigned int*)h_bf,  (long long)BB * HH / 2);
  zero_u32<<<2048, 256, 0, stream>>>((unsigned int*)y_bf0, (long long)BB * OUT_DIM / 2);

  // xW1b = x @ W1 + b1   (loop-invariant drive)
  gemm_bf16_wmma<0><<<dim3(HH / BN, BB / BM), 256, 0, stream>>>(
      x_bf, W1t, BB, HH, IN_DIM, b1, nullptr, nullptr, nullptr, xW1b, nullptr);

  __bf16* ycur = y_bf0;
  __bf16* ynxt = y_bf1;
  for (int t = 0; t < T_RELAX; ++t) {
    // y_{t+1} = clip(EPS*(h_t @ W2 + b2) + CT*target)   [y_t coeff == 0]
    gemm_bf16_wmma<2><<<dim3(OUT_DIM / BN, BB / BM), 256, 0, stream>>>(
        h_bf, W2t, BB, OUT_DIM, HH, b2, nullptr, nullptr, tgt, y_out, ynxt);
    // h_{t+1} = clip(CH*h_t + EPS*(xW1b + y_t @ W2^T))
    float* ho = (t == T_RELAX - 1) ? h_out : h_f32;
    gemm_bf16_wmma<1><<<dim3(HH / BN, BB / BM), 256, 0, stream>>>(
        ycur, W2bf, BB, HH, OUT_DIM, nullptr, xW1b, h_f32, nullptr, ho, h_bf);
    __bf16* tmp = ycur; ycur = ynxt; ynxt = tmp;
  }

  (void)in_sizes; (void)n_in; (void)out_size; (void)ws_size;
}
